// Attention_65764539236808
// MI455X (gfx1250) — matrix operand order repaired and checked by reading, not compiled, not run on hardware
//
#include <hip/hip_runtime.h>

#define SEQ      2048
#define BATCH    2
#define HIDDEN   2048
#define HEADS    16
#define HEAD_DIM 128
#define O3       (3 * HIDDEN)      // 6144
#define MROWS    (SEQ * BATCH)     // 4096

typedef __attribute__((ext_vector_type(16))) _Float16 v16h;
typedef __attribute__((ext_vector_type(8)))  _Float16 v8h;
typedef __attribute__((ext_vector_type(8)))  float    v8f;

static __device__ inline v16h frag2(const _Float16* p0, const _Float16* p1) {
    v8h a = *(const v8h*)p0;
    v8h b = *(const v8h*)p1;
    return __builtin_shufflevector(a, b, 0,1,2,3,4,5,6,7,8,9,10,11,12,13,14,15);
}

static __device__ inline v8f wmma16(v16h a, v16h b, v8f c) {
    // D = A(16x32 f16) * B(32x16 f16) + C(16x16 f32)
    return __builtin_amdgcn_wmma_f32_16x16x32_f16(false, a, false, b, (short)0, c, false, false);
}

// Async DMA: copy 32 bytes global -> LDS per lane (two B128 transfers).
// ISA 10.x / 15.18: INST_OFFSET is added to BOTH the LDS (VDST) and global
// (VADDR) addresses, so one address pair + offset:16 covers both halves.
// LDS byte address = low 32 bits of the generic pointer (aperture split).
static __device__ inline void async_copy32(const void* gaddr, const void* lds_generic) {
    unsigned lds0 = (unsigned)(unsigned long long)lds_generic;
    asm volatile("global_load_async_to_lds_b128 %0, %1, off\n\t"
                 "global_load_async_to_lds_b128 %0, %1, off offset:16"
                 :: "v"(lds0), "v"(gaddr) : "memory");
}
static __device__ inline void async_wait0() {
    asm volatile("s_wait_asynccnt 0x0" ::: "memory");
}

static __device__ inline float rowmax16(float v) {
#pragma unroll
    for (int off = 8; off > 0; off >>= 1) v = fmaxf(v, __shfl_xor(v, off, 32));
    return v;
}
static __device__ inline float rowsum16(float v) {
#pragma unroll
    for (int off = 8; off > 0; off >>= 1) v += __shfl_xor(v, off, 32);
    return v;
}

// ---------------------------------------------------------------------------
// C[M,N] = A[M,K] * W[N,K]^T + bias[N]
// A: fp32 (A_HALF=0, converted during staging) or f16 (A_HALF=1, async DMA).
// W: fp32 row-major [N][K]. Out: f16 (OUT_HALF=1) or fp32.
// Tiles: BM=BN=128, BK=32; 8 waves as 2x4, each wave = 4x2 WMMA tiles.
// ---------------------------------------------------------------------------
template<bool A_HALF, bool OUT_HALF>
__global__ __launch_bounds__(256) void gemm_bias_wmma(
    const void* __restrict__ Av, const float* __restrict__ Bw,
    const float* __restrict__ bias, void* __restrict__ Cv,
    int M, int N, int K) {
    __shared__ __align__(16) _Float16 As[128][32];
    __shared__ __align__(16) _Float16 Bs[128][32];

    const int tid   = threadIdx.x;
    const int lane  = tid & 31, wave = tid >> 5;
    const int waveM = wave >> 2, waveN = wave & 3;   // 2 x 4 wave grid
    const int lr    = lane & 15;
    const int hlf   = lane >> 4;
    const int m_blk = blockIdx.y * 128;
    const int n_blk = blockIdx.x * 128;
    const int sr    = tid >> 1;          // staging row 0..127
    const int sc    = (tid & 1) * 16;    // staging col 0 / 16

    const v8f z8 = {0.f,0.f,0.f,0.f,0.f,0.f,0.f,0.f};
    v8f acc[4][2];
#pragma unroll
    for (int i = 0; i < 4; ++i)
#pragma unroll
        for (int j = 0; j < 2; ++j) acc[i][j] = z8;

    for (int k0 = 0; k0 < K; k0 += 32) {
        __syncthreads();
        // ---- stage A tile ----
        if (A_HALF) {
            const _Float16* A = (const _Float16*)Av;
            const _Float16* p = A + (size_t)(m_blk + sr) * K + (k0 + sc);
            async_copy32(p, &As[sr][sc]);      // DMA, tracked by ASYNCcnt
        } else {
            const float* A = (const float*)Av;
            const float* p = A + (size_t)(m_blk + sr) * K + (k0 + sc);
            float4 f0 = *(const float4*)(p + 0);
            float4 f1 = *(const float4*)(p + 4);
            float4 f2 = *(const float4*)(p + 8);
            float4 f3 = *(const float4*)(p + 12);
            v8h h0 = {(_Float16)f0.x,(_Float16)f0.y,(_Float16)f0.z,(_Float16)f0.w,
                      (_Float16)f1.x,(_Float16)f1.y,(_Float16)f1.z,(_Float16)f1.w};
            v8h h1 = {(_Float16)f2.x,(_Float16)f2.y,(_Float16)f2.z,(_Float16)f2.w,
                      (_Float16)f3.x,(_Float16)f3.y,(_Float16)f3.z,(_Float16)f3.w};
            *(v8h*)&As[sr][sc]     = h0;
            *(v8h*)&As[sr][sc + 8] = h1;
        }
        // ---- stage B tile (weights, fp32 -> f16) ----
        {
            const float* p = Bw + (size_t)(n_blk + sr) * K + (k0 + sc);
            float4 f0 = *(const float4*)(p + 0);
            float4 f1 = *(const float4*)(p + 4);
            float4 f2 = *(const float4*)(p + 8);
            float4 f3 = *(const float4*)(p + 12);
            v8h h0 = {(_Float16)f0.x,(_Float16)f0.y,(_Float16)f0.z,(_Float16)f0.w,
                      (_Float16)f1.x,(_Float16)f1.y,(_Float16)f1.z,(_Float16)f1.w};
            v8h h1 = {(_Float16)f2.x,(_Float16)f2.y,(_Float16)f2.z,(_Float16)f2.w,
                      (_Float16)f3.x,(_Float16)f3.y,(_Float16)f3.z,(_Float16)f3.w};
            *(v8h*)&Bs[sr][sc]     = h0;
            *(v8h*)&Bs[sr][sc + 8] = h1;
        }
        if (A_HALF) async_wait0();             // DMA'd bytes visible in LDS
        __syncthreads();

        v16h af[4], bf[2];
#pragma unroll
        for (int i = 0; i < 4; ++i) {
            const _Float16* rp = &As[waveM * 64 + i * 16 + lr][0];
            af[i] = frag2(rp + hlf * 8, rp + 16 + hlf * 8);        // A layout
        }
#pragma unroll
        for (int j = 0; j < 2; ++j) {
            const _Float16* rp = &Bs[waveN * 32 + j * 16 + lr][0];
            bf[j] = frag2(rp + hlf * 8, rp + 16 + hlf * 8);        // B layout
        }
#pragma unroll
        for (int i = 0; i < 4; ++i)
#pragma unroll
            for (int j = 0; j < 2; ++j)
                acc[i][j] = wmma16(af[i], bf[j], acc[i][j]);
    }

    // ---- epilogue: bias + store ----
#pragma unroll
    for (int i = 0; i < 4; ++i) {
#pragma unroll
        for (int j = 0; j < 2; ++j) {
            const int nCol = n_blk + waveN * 32 + j * 16 + lr;
            const float bv = bias[nCol];
#pragma unroll
            for (int r = 0; r < 8; ++r) {
                const int mRow = m_blk + waveM * 64 + i * 16 + r + hlf * 8;
                const float v = acc[i][j][r] + bv;
                if (OUT_HALF) ((_Float16*)Cv)[(size_t)mRow * N + nCol] = (_Float16)v;
                else          ((float*)   Cv)[(size_t)mRow * N + nCol] = v;
            }
        }
    }
}

// ---------------------------------------------------------------------------
// Flash attention, causal. qkv: f16 [SEQ][BATCH][6144] where per head n the
// columns n*384+{0,128,256} are Q/K/V. ctx out: f16 [SEQ][BATCH][HIDDEN].
// Block = 128 queries of one (batch, head); wave = 16 queries.
// K tiles are DMA'd to LDS (async); V is transposed through VGPRs.
// ---------------------------------------------------------------------------
__global__ __launch_bounds__(256) void attn_wmma(
    const _Float16* __restrict__ qkv, _Float16* __restrict__ ctx) {
    __shared__ __align__(16) _Float16 Ks[32][136];   // raw K block (key, dim)
    __shared__ __align__(16) _Float16 Vt[128][40];   // transposed V (dim, key)
    __shared__ __align__(16) _Float16 Ps[8][16][32]; // per-wave P transpose patch

    const int tid   = threadIdx.x;
    const int lane  = tid & 31, wave = tid >> 5;
    const int lr    = lane & 15, hlf = lane >> 4;
    const int qb    = blockIdx.x;
    const int batch = blockIdx.y >> 4;
    const int head  = blockIdx.y & 15;
    const int q0    = qb * 128 + wave * 16;
    const float scale = 0.08838834764831845f;        // 1/sqrt(128)

    // Q fragments: 16 rows x 128 dims = 4 A-fragments, resident all kernel.
    v16h qf[4];
    {
        const _Float16* Qb = qkv + ((size_t)(q0 + lr) * BATCH + batch) * O3 + head * 384;
#pragma unroll
        for (int kk = 0; kk < 4; ++kk) {
            const _Float16* p = Qb + kk * 32 + hlf * 8;
            qf[kk] = frag2(p, p + 16);
        }
    }

    const v8f z8 = {0.f,0.f,0.f,0.f,0.f,0.f,0.f,0.f};
    v8f Ot[8];
    float m[8], l[8];
#pragma unroll
    for (int j = 0; j < 8; ++j) Ot[j] = z8;
#pragma unroll
    for (int r = 0; r < 8; ++r) { m[r] = -3.0e38f; l[r] = 0.f; }

    const int skey = tid >> 3;        // staging: key 0..31
    const int sd   = (tid & 7) * 16;  // staging: dim 0..112

    const int nkb = qb * 4 + 4;       // causal bound for this query block
    for (int kb = 0; kb < nkb; ++kb) {
        const int t0 = kb * 32;
        __syncthreads();
        // ---- stage K via async DMA; V loaded + transposed through VGPRs ----
        {
            const _Float16* Kp = qkv + ((size_t)(t0 + skey) * BATCH + batch) * O3
                                 + head * 384 + 128 + sd;
            async_copy32(Kp, &Ks[skey][sd]);
            const _Float16* Vp = qkv + ((size_t)(t0 + skey) * BATCH + batch) * O3
                                 + head * 384 + 256 + sd;
            v8h v0 = *(const v8h*)Vp;
            v8h v1 = *(const v8h*)(Vp + 8);
#pragma unroll
            for (int e = 0; e < 8; ++e) {
                Vt[sd + e][skey]     = v0[e];
                Vt[sd + 8 + e][skey] = v1[e];
            }
        }
        async_wait0();
        __syncthreads();

        // ---- S = Q * K^T : two 16x16 tiles over 32 keys ----
        v8f s0 = z8, s1 = z8;
#pragma unroll
        for (int kk = 0; kk < 4; ++kk) {
            const _Float16* r0 = &Ks[lr][0];
            const _Float16* r1 = &Ks[16 + lr][0];
            v16h b0 = frag2(r0 + kk * 32 + hlf * 8, r0 + kk * 32 + 16 + hlf * 8);
            v16h b1 = frag2(r1 + kk * 32 + hlf * 8, r1 + kk * 32 + 16 + hlf * 8);
            s0 = wmma16(qf[kk], b0, s0);
            s1 = wmma16(qf[kk], b1, s1);
        }

        // ---- causal mask + online softmax ----
        float alpha[8];
#pragma unroll
        for (int r = 0; r < 8; ++r) {
            const int gq = q0 + r + hlf * 8;
            const float sv0 = (t0 + lr      <= gq) ? s0[r] * scale : -3.0e38f;
            const float sv1 = (t0 + 16 + lr <= gq) ? s1[r] * scale : -3.0e38f;
            const float rm = rowmax16(fmaxf(sv0, sv1));
            const float nm = fmaxf(m[r], rm);
            const float al = __expf(m[r] - nm);
            const float p0 = __expf(sv0 - nm);
            const float p1 = __expf(sv1 - nm);
            const float rs = rowsum16(p0 + p1);
            l[r] = l[r] * al + rs;
            m[r] = nm;
            alpha[r] = al;
            s0[r] = p0;
            s1[r] = p1;
        }
#pragma unroll
        for (int j = 0; j < 8; ++j)
#pragma unroll
            for (int r = 0; r < 8; ++r) Ot[j][r] *= alpha[r];

        // ---- transpose P (C layout -> A layout) through per-wave LDS ----
#pragma unroll
        for (int r = 0; r < 8; ++r) {
            const int row = r + hlf * 8;
            Ps[wave][row][lr]      = (_Float16)s0[r];
            Ps[wave][row][16 + lr] = (_Float16)s1[r];
        }
        __syncthreads();
        v16h pf;
        {
            const _Float16* pp = &Ps[wave][lr][0];
            pf = frag2(pp + hlf * 8, pp + 16 + hlf * 8);
        }
        // ---- O += P * V over all 128 dims ----
#pragma unroll
        for (int j = 0; j < 8; ++j) {
            const _Float16* vp = &Vt[j * 16 + lr][0];
            v16h vf = frag2(vp + hlf * 8, vp + 16 + hlf * 8);
            Ot[j] = wmma16(pf, vf, Ot[j]);
        }
    }

    // ---- normalize and store context (f16) ----
#pragma unroll
    for (int j = 0; j < 8; ++j) {
#pragma unroll
        for (int r = 0; r < 8; ++r) {
            const int row = q0 + r + hlf * 8;
            const int d   = j * 16 + lr;
            const float v = Ot[j][r] / l[r];
            ctx[((size_t)row * BATCH + batch) * HIDDEN + head * HEAD_DIM + d] = (_Float16)v;
        }
    }
}

// ---------------------------------------------------------------------------
extern "C" void kernel_launch(void* const* d_in, const int* in_sizes, int n_in,
                              void* d_out, int out_size, void* d_ws, size_t ws_size,
                              hipStream_t stream) {
    (void)in_sizes; (void)n_in; (void)out_size; (void)ws_size;
    const float* x       = (const float*)d_in[0];
    // d_in[1] is the boolean mask; causality is computed analytically.
    const float* w_qkv   = (const float*)d_in[2];
    const float* b_qkv   = (const float*)d_in[3];
    const float* w_dense = (const float*)d_in[4];
    const float* b_dense = (const float*)d_in[5];
    float* out = (float*)d_out;

    _Float16* qkvh = (_Float16*)d_ws;                       // [4096][6144] f16
    _Float16* ctxh = qkvh + (size_t)MROWS * O3;             // [4096][2048] f16

    // 1) QKV projection: [4096,2048] x [6144,2048]^T + b_qkv -> f16
    gemm_bias_wmma<false, true><<<dim3(O3 / 128, MROWS / 128), 256, 0, stream>>>(
        x, w_qkv, b_qkv, qkvh, MROWS, O3, HIDDEN);

    // 2) causal flash attention per (batch, head)
    attn_wmma<<<dim3(SEQ / 128, BATCH * HEADS), 256, 0, stream>>>(qkvh, ctxh);

    // 3) dense projection: [4096,2048] x [2048,2048]^T + b_dense -> fp32 out
    gemm_bias_wmma<true, false><<<dim3(HIDDEN / 128, MROWS / 128), 256, 0, stream>>>(
        ctxh, w_dense, b_dense, out, MROWS, HIDDEN, HIDDEN);
}
